// TransformerDecoderLayerQaN_64879775973736
// MI455X (gfx1250) — matrix operand order repaired and checked by reading, not compiled, not run on hardware
//
#include <hip/hip_runtime.h>
#include <hip/hip_bf16.h>
#include <math.h>

// ---------------------------------------------------------------------------
// TransformerDecoderLayerQaN for MI455X (gfx1250, wave32, WMMA).
// Heavy GEMMs: bf16 inputs, f32 accumulate via v_wmma_f32_16x16x32_bf16.
// Weight tiles double-buffered in LDS via GLOBAL_LOAD_ASYNC_TO_LDS (ASYNCcnt),
// overlapping the DMA for tile i+1 with the WMMA work on tile i.
// ---------------------------------------------------------------------------

#define T_DIM 2048
#define B_DIM 8
#define S_DIM 512
#define D_DIM 512
#define H_DIM 8
#define DH    64
#define DFF_D 2048
#define NQ_D  10
#define WIN   16
#define NW    (T_DIM / WIN)
#define EPSLN 1e-5f
#define NEGBIG -1e30f

typedef __attribute__((ext_vector_type(16))) __bf16 v16bf;
typedef __attribute__((ext_vector_type(8)))  float  v8f;

union FragU { v16bf v; uint4 q[2]; };

__device__ inline v8f zero_v8f() {
  v8f z;
#pragma unroll
  for (int i = 0; i < 8; ++i) z[i] = 0.0f;
  return z;
}

__device__ inline v8f wmma_bf16(v16bf a, v16bf b, v8f c) {
  // (neg_a, A, neg_b, B, c_mod, C, reuse_a, reuse_b)
  return __builtin_amdgcn_wmma_f32_16x16x32_bf16(false, a, false, b, (short)0, c,
                                                 false, false);
}

// Async memory->LDS copy, tracked by ASYNCcnt (CDNA5 ISA).
// VDST = LDS byte address (low 32 bits of the generic shared pointer),
// VADDR = 64-bit global address, GV mode (saddr = off).
__device__ inline void async_load_b128_to_lds(void* lds_ptr, const void* gptr) {
  unsigned lds_addr = (unsigned)(unsigned long long)lds_ptr;
  asm volatile("global_load_async_to_lds_b128 %0, %1, off"
               :
               : "v"(lds_addr), "v"(gptr)
               : "memory");
}
// Wait until at most N async ops outstanding.  Async loads complete in order,
// so N==1 with two in flight means the oldest tile has fully landed in LDS.
__device__ inline void wait_asynccnt0() {
  asm volatile("s_wait_asynccnt 0x0" ::: "memory");
}
__device__ inline void wait_asynccnt1() {
  asm volatile("s_wait_asynccnt 0x1" ::: "memory");
}

// A fragment (16x32 bf16). Lanes 0-15 hold row M=lane with
// K = {k0..k0+7, k0+16..k0+23}; lanes 16-31 hold row M=lane-16 with
// K = {k0+8..k0+15, k0+24..k0+31}.  Two 16B loads per lane.
__device__ inline v16bf load_a_frag(const __bf16* base, long long row_stride,
                                    int row0, int k0, int lane) {
  const int half = lane >> 4;
  const int r = row0 + (lane & 15);
  const __bf16* p = base + (long long)r * row_stride + k0 + half * 8;
  FragU u;
  u.q[0] = *(const uint4*)(p);
  u.q[1] = *(const uint4*)(p + 16);
  return u.v;
}

// B fragment (32x16 bf16). Lanes 0-15: column N=lane, K={k0..k0+7, k0+16..k0+23};
// lanes 16-31: column N=lane-16, K={k0+8..k0+15, k0+24..k0+31}.  Two 16B loads per lane.
__device__ inline v16bf load_b_frag(const __bf16* base, long long col_stride,
                                    int n0, int k0, int lane) {
  const int n = n0 + (lane & 15);
  const int k = k0 + (lane >> 4) * 8;
  const __bf16* p = base + (long long)n * col_stride + k;
  FragU u;
  u.q[0] = *(const uint4*)(p);
  u.q[1] = *(const uint4*)(p + 16);
  return u.v;
}

// ---------------------------------------------------------------------------
// Generic GEMM: C[M,N] = A[M,K](bf16) * W[N,K]^T(bf16) + bias, opt ReLU.
// 256 threads = 8 waves; each wave owns 16 rows x 64 cols (4 accumulators);
// workgroup tile = 128 rows x 64 cols.  The 64n x 32k weight tile is double-
// buffered in LDS, fed by ASYNCcnt-tracked async loads so the DMA for tile
// i+1 overlaps the WMMAs on tile i; reused by all 8 waves.
// ---------------------------------------------------------------------------
__global__ void __launch_bounds__(256) gemm_bf16_kernel(
    const __bf16* __restrict__ A, const __bf16* __restrict__ W,
    const float* __restrict__ bias, float* __restrict__ Cf,
    __bf16* __restrict__ Cb, int M, int N, int K, int relu) {
  __shared__ __bf16 Bs[2][64 * 48];  // 48-elem row stride: 16B aligned
  const int tid  = threadIdx.x;
  const int lane = tid & 31;
  const int wave = tid >> 5;
  const int mBase = blockIdx.x * 128;
  const int nBase = blockIdx.y * 64;
  const int mRow = mBase + wave * 16;

  v8f acc[4];
#pragma unroll
  for (int i = 0; i < 4; ++i) acc[i] = zero_v8f();

  // each of the 256 threads owns one 16B chunk of the 64x32 weight tile
  const int sr = tid >> 2;          // row 0..63
  const int sc = (tid & 3) * 8;     // k-chunk {0,8,16,24}
  const __bf16* wrow = W + (long long)(nBase + sr) * K + sc;

  const int nk = K >> 5;
  // prologue: kick off tile 0
  async_load_b128_to_lds(&Bs[0][sr * 48 + sc], wrow);

  for (int ki = 0; ki < nk; ++ki) {
    const int cur = ki & 1;
    if (ki + 1 < nk) {
      // issue DMA for the next tile into the other buffer, then wait only
      // for the oldest copy (this tile) to land.
      async_load_b128_to_lds(&Bs[cur ^ 1][sr * 48 + sc],
                             wrow + (ki + 1) * 32);
      wait_asynccnt1();
    } else {
      wait_asynccnt0();
    }
    __syncthreads();

    const int k0 = ki * 32;
    v16bf a = load_a_frag(A, K, mRow, k0, lane);
    if (ki + 1 < nk)
      __builtin_prefetch(A + (long long)(mRow + (lane & 15)) * K + k0 + 32, 0, 0);

    const __bf16* bsrc = &Bs[cur][0];
#pragma unroll
    for (int i = 0; i < 4; ++i) {
      v16bf b = load_b_frag(bsrc, 48, i * 16, 0, lane);
      acc[i] = wmma_bf16(a, b, acc[i]);
    }
    __syncthreads();  // all waves done reading Bs[cur] before it is refilled
  }

  // C/D layout: vgpr j, lanes 0-15 -> (M=j, N=lane); lanes 16-31 -> (M=j+8).
  const int rb = mRow + (lane >> 4) * 8;
#pragma unroll
  for (int i = 0; i < 4; ++i) {
    const int n = nBase + i * 16 + (lane & 15);
    const float bv = bias ? bias[n] : 0.0f;
#pragma unroll
    for (int j = 0; j < 8; ++j) {
      float v = acc[i][j] + bv;
      if (relu) v = fmaxf(v, 0.0f);
      const long long idx = (long long)(rb + j) * N + n;
      if (Cf) Cf[idx] = v;
      if (Cb) Cb[idx] = (__bf16)v;
    }
  }
}

// ---------------------------------------------------------------------------
// Fused cross-attention: per (b,h) and 64-row T tile.
//   scores(64x512) = Q K^T / 8  (WMMA, staged in LDS f32)
//   softmax over full S=512 (fits in LDS; no streaming stats needed)
//   O(64x64) = P V             (WMMA, P converted f32->bf16 from LDS)
// Dynamic LDS ~134 KB; WGP has 320 KB so 2 blocks/WGP still fit.
// ---------------------------------------------------------------------------
__global__ void __launch_bounds__(256) attn_kernel(
    const __bf16* __restrict__ Qb, const __bf16* __restrict__ Kb,
    const __bf16* __restrict__ VT, __bf16* __restrict__ Ob) {
  extern __shared__ char smem[];
  float* sc  = (float*)smem;                       // [64][516]
  float* red = (float*)(smem + 64 * 516 * 4);      // [64][8]
  const int LD = 516;
  const int tid  = threadIdx.x;
  const int lane = tid & 31;
  const int wave = tid >> 5;
  const int ttile = blockIdx.x;       // T/64 tiles
  const int bh    = blockIdx.y;       // B*H
  const int b = bh / H_DIM, h = bh % H_DIM;
  const long long TOKS = (long long)B_DIM * D_DIM;   // token row stride
  const __bf16* Qbase = Qb + (long long)b * D_DIM + h * DH;
  const __bf16* Kbase = Kb + (long long)b * D_DIM + h * DH;

  // ---- Phase 1: scores = (Q K^T) * dh^-0.5 ----
  {
    const int rtile  = wave & 3;       // 4 row tiles of 16
    const int cgroup = wave >> 2;      // 2 col groups of 256
    const int row0 = ttile * 64 + rtile * 16;
    v16bf aq0 = load_a_frag(Qbase, TOKS, row0, 0,  lane);
    v16bf aq1 = load_a_frag(Qbase, TOKS, row0, 32, lane);
#pragma unroll 4
    for (int ct = 0; ct < 16; ++ct) {
      const int s0 = (cgroup * 16 + ct) * 16;
      v8f acc = zero_v8f();
      v16bf bk0 = load_b_frag(Kbase, TOKS, s0, 0, lane);
      acc = wmma_bf16(aq0, bk0, acc);
      v16bf bk1 = load_b_frag(Kbase, TOKS, s0, 32, lane);
      acc = wmma_bf16(aq1, bk1, acc);
      const int col = s0 + (lane & 15);
      const int rb = rtile * 16 + (lane >> 4) * 8;
#pragma unroll
      for (int j = 0; j < 8; ++j) sc[(rb + j) * LD + col] = acc[j] * 0.125f;
    }
  }
  __syncthreads();

  // ---- Phase 2: row softmax over S=512 (4 threads per row) ----
  {
    const int row = tid >> 2, part = tid & 3;
    float* rp = &sc[row * LD + part * 128];
    float lm = -INFINITY;
    for (int c = 0; c < 128; ++c) lm = fmaxf(lm, rp[c]);
    red[row * 8 + part] = lm;
    __syncthreads();
    const float m = fmaxf(fmaxf(red[row * 8 + 0], red[row * 8 + 1]),
                          fmaxf(red[row * 8 + 2], red[row * 8 + 3]));
    float ls = 0.0f;
    for (int c = 0; c < 128; ++c) {
      float e = __expf(rp[c] - m);
      rp[c] = e;
      ls += e;
    }
    red[row * 8 + 4 + part] = ls;
    __syncthreads();
    const float s = red[row * 8 + 4] + red[row * 8 + 5] +
                    red[row * 8 + 6] + red[row * 8 + 7];
    const float inv = 1.0f / s;
    for (int c = 0; c < 128; ++c) rp[c] *= inv;
  }
  __syncthreads();

  // ---- Phase 3: O = P V  (P from LDS f32 -> bf16, V^T contiguous) ----
  {
    const int rtile = wave >> 1;       // 4 row tiles
    const int cpair = wave & 1;        // 2 x (2 col tiles of 16)
    const __bf16* Vbase = VT + (long long)bh * DH * S_DIM;
    v8f acc0 = zero_v8f(), acc1 = zero_v8f();
    const int half = lane >> 4;
    const int r = rtile * 16 + (lane & 15);
    for (int ks = 0; ks < 16; ++ks) {
      const int s0 = ks * 32;
      FragU u;
      const float* sp = &sc[r * LD + s0 + half * 8];
#pragma unroll
      for (int i2 = 0; i2 < 8; ++i2) {
        u.v[i2]     = (__bf16)sp[i2];
        u.v[8 + i2] = (__bf16)sp[16 + i2];
      }
      v16bf bv0 = load_b_frag(Vbase, S_DIM, cpair * 32,      s0, lane);
      acc0 = wmma_bf16(u.v, bv0, acc0);
      v16bf bv1 = load_b_frag(Vbase, S_DIM, cpair * 32 + 16, s0, lane);
      acc1 = wmma_bf16(u.v, bv1, acc1);
    }
    const int rb = ttile * 64 + rtile * 16 + (lane >> 4) * 8;
    const int d0 = cpair * 32 + (lane & 15);
#pragma unroll
    for (int j = 0; j < 8; ++j) {
      const long long o = ((long long)(rb + j) * B_DIM + b) * D_DIM + h * DH;
      Ob[o + d0]      = (__bf16)acc0[j];
      Ob[o + d0 + 16] = (__bf16)acc1[j];
    }
  }
}

// ---------------------------------------------------------------------------
// Normalize learned queries: per (n,h) unit-norm, then 1/sqrt(dh) and the
// local-attn scale 1/sqrt(D) folded in.
// ---------------------------------------------------------------------------
__global__ void __launch_bounds__(64) prep_q_kernel(
    const float* __restrict__ queries, float* __restrict__ qn) {
  __shared__ float red[64];
  const int n = blockIdx.x / H_DIM, h = blockIdx.x % H_DIM;
  const int i = threadIdx.x;
  const float v = queries[n * D_DIM + h * DH + i];
  red[i] = v * v;
  __syncthreads();
  for (int off = 32; off > 0; off >>= 1) {
    if (i < off) red[i] += red[i + off];
    __syncthreads();
  }
  const float nrm = sqrtf(red[0]);
  qn[n * D_DIM + h * DH + i] = v / (nrm + 1e-6f) * (0.125f * 0.04419417382415922f);
}

// ---------------------------------------------------------------------------
// QA local-attention block fused with residual-add + LN1.
// One block per (window m, batch b).  Algebraic collapse: output of a window
// is sum_j w[j]*kv[j,:] with w[j] = sum_n wk[n]*softmax(sim)[n,j].
// ---------------------------------------------------------------------------
__global__ void __launch_bounds__(256) qa_ln1_kernel(
    const float* __restrict__ tgt, const float* __restrict__ qn,
    const float* __restrict__ wkv, const float* __restrict__ g,
    const float* __restrict__ be, float* __restrict__ x1f,
    __bf16* __restrict__ x1b) {
  extern __shared__ char smem[];
  float* kv  = (float*)smem;           // [48][512]
  float* sim = kv + 48 * D_DIM;        // [10][48]
  float* wj  = sim + NQ_D * 48;        // [48]
  float* qa  = wj + 48;                // [512]
  float* red = qa + D_DIM;             // [256]
  const int tid = threadIdx.x;
  const int m = blockIdx.x, b = blockIdx.y;

  for (int idx = tid; idx < 48 * D_DIM; idx += 256) {
    const int j = idx >> 9, d = idx & 511;
    const int t = (m - 1) * WIN + j;
    kv[idx] = (t >= 0 && t < T_DIM)
                  ? tgt[((long long)t * B_DIM + b) * D_DIM + d]
                  : 0.0f;
  }
  __syncthreads();

  for (int e = tid; e < NQ_D * 48; e += 256) {
    const int n = e / 48, j = e % 48;
    const int w = j / WIN;
    const bool ok = (w == 1) || (w == 0 && m > 0) || (w == 2 && m < NW - 1);
    float sv = NEGBIG;
    if (ok) {
      const float* qp = qn + n * D_DIM;
      const float* kp = kv + j * D_DIM;
      float s = 0.0f;
      for (int d = 0; d < D_DIM; ++d) s += qp[d] * kp[d];
      sv = s;
    }
    sim[e] = sv;
  }
  __syncthreads();

  if (tid < NQ_D) {
    float* row = sim + tid * 48;
    float mx = -INFINITY;
    for (int j = 0; j < 48; ++j) mx = fmaxf(mx, row[j]);
    float s = 0.0f;
    for (int j = 0; j < 48; ++j) { float e = __expf(row[j] - mx); row[j] = e; s += e; }
    const float inv = 1.0f / s;
    for (int j = 0; j < 48; ++j) row[j] *= inv;
  }
  __syncthreads();

  if (tid < 48) {
    float s = 0.0f;
    for (int n = 0; n < NQ_D; ++n) s += wkv[n] * sim[n * 48 + tid];
    wj[tid] = s;
  }
  __syncthreads();

  for (int d = tid; d < D_DIM; d += 256) {
    float s = 0.0f;
    for (int j = 0; j < 48; ++j) s += wj[j] * kv[j * D_DIM + d];
    qa[d] = s;
  }
  __syncthreads();

  for (int p = 0; p < WIN; ++p) {
    const int t = m * WIN + p;
    const float* xr = tgt + ((long long)t * B_DIM + b) * D_DIM;
    const int d0 = tid * 2;
    const float v0 = xr[d0] + qa[d0];
    const float v1 = xr[d0 + 1] + qa[d0 + 1];
    red[tid] = v0 + v1;
    __syncthreads();
    for (int off = 128; off > 0; off >>= 1) {
      if (tid < off) red[tid] += red[tid + off];
      __syncthreads();
    }
    const float mean = red[0] * (1.0f / D_DIM);
    __syncthreads();
    const float c0 = v0 - mean, c1 = v1 - mean;
    red[tid] = c0 * c0 + c1 * c1;
    __syncthreads();
    for (int off = 128; off > 0; off >>= 1) {
      if (tid < off) red[tid] += red[tid + off];
      __syncthreads();
    }
    const float var = red[0] * (1.0f / D_DIM);
    __syncthreads();
    const float rs = rsqrtf(var + EPSLN);
    const float y0 = c0 * rs * g[d0] + be[d0];
    const float y1 = c1 * rs * g[d0 + 1] + be[d0 + 1];
    const long long o = ((long long)t * B_DIM + b) * D_DIM + d0;
    x1f[o] = y0;     x1f[o + 1] = y1;
    x1b[o] = (__bf16)y0; x1b[o + 1] = (__bf16)y1;
  }
}

// ---------------------------------------------------------------------------
// out = LN(x + delta); f32 out always, optional bf16 mirror.
// ---------------------------------------------------------------------------
__global__ void __launch_bounds__(256) add_ln_kernel(
    const float* __restrict__ x, const float* __restrict__ dlt,
    const float* __restrict__ g, const float* __restrict__ be,
    float* __restrict__ outf, __bf16* __restrict__ outb) {
  __shared__ float red[256];
  const long long row = blockIdx.x;
  const int tid = threadIdx.x, d0 = tid * 2;
  const float* xr = x + row * D_DIM;
  const float* dr = dlt + row * D_DIM;
  const float v0 = xr[d0] + dr[d0];
  const float v1 = xr[d0 + 1] + dr[d0 + 1];
  red[tid] = v0 + v1;
  __syncthreads();
  for (int off = 128; off > 0; off >>= 1) {
    if (tid < off) red[tid] += red[tid + off];
    __syncthreads();
  }
  const float mean = red[0] * (1.0f / D_DIM);
  __syncthreads();
  const float c0 = v0 - mean, c1 = v1 - mean;
  red[tid] = c0 * c0 + c1 * c1;
  __syncthreads();
  for (int off = 128; off > 0; off >>= 1) {
    if (tid < off) red[tid] += red[tid + off];
    __syncthreads();
  }
  const float var = red[0] * (1.0f / D_DIM);
  const float rs = rsqrtf(var + EPSLN);
  const float y0 = c0 * rs * g[d0] + be[d0];
  const float y1 = c1 * rs * g[d0 + 1] + be[d0 + 1];
  const long long o = row * D_DIM + d0;
  outf[o] = y0;
  outf[o + 1] = y1;
  if (outb) { outb[o] = (__bf16)y0; outb[o + 1] = (__bf16)y1; }
}

// f32 -> bf16 grid-stride convert
__global__ void cvt_bf16_kernel(const float* __restrict__ src,
                                __bf16* __restrict__ dst, long long n) {
  long long i = (long long)blockIdx.x * blockDim.x + threadIdx.x;
  const long long stride = (long long)gridDim.x * blockDim.x;
  for (; i < n; i += stride) dst[i] = (__bf16)src[i];
}

// V[(s*B+b)*D + h*64+d] -> VT[((b*H+h)*64+d)*S + s]
__global__ void vt_kernel(const __bf16* __restrict__ Vb, __bf16* __restrict__ VT) {
  const long long total = (long long)B_DIM * H_DIM * DH * S_DIM;
  long long idx = (long long)blockIdx.x * blockDim.x + threadIdx.x;
  if (idx >= total) return;
  const int s   = (int)(idx & (S_DIM - 1));
  const int rem = (int)(idx >> 9);
  const int d   = rem & (DH - 1);
  const int bh  = rem >> 6;
  const int b = bh / H_DIM, h = bh % H_DIM;
  VT[idx] = Vb[((long long)s * B_DIM + b) * D_DIM + h * DH + d];
}

// ---------------------------------------------------------------------------
extern "C" void kernel_launch(void* const* d_in, const int* in_sizes, int n_in,
                              void* d_out, int out_size, void* d_ws,
                              size_t ws_size, hipStream_t stream) {
  (void)in_sizes; (void)n_in; (void)out_size; (void)ws_size;
  const float* tgt     = (const float*)d_in[0];
  const float* memory  = (const float*)d_in[1];
  const float* queries = (const float*)d_in[2];
  const float* wk      = (const float*)d_in[3];
  const float* in_w    = (const float*)d_in[4];
  const float* in_b    = (const float*)d_in[5];
  const float* out_w   = (const float*)d_in[6];
  const float* out_b   = (const float*)d_in[7];
  const float* l1w     = (const float*)d_in[8];
  const float* l1b     = (const float*)d_in[9];
  const float* l2w     = (const float*)d_in[10];
  const float* l2b     = (const float*)d_in[11];
  const float* ln1g    = (const float*)d_in[12];
  const float* ln1b    = (const float*)d_in[13];
  const float* ln2g    = (const float*)d_in[14];
  const float* ln2b    = (const float*)d_in[15];
  const float* ln3g    = (const float*)d_in[16];
  const float* ln3b    = (const float*)d_in[17];

  char* ws = (char*)d_ws;
  size_t off = 0;
  auto alloc = [&](size_t bytes) -> void* {
    void* p = ws + off;
    off = (off + bytes + 255) & ~(size_t)255;
    return p;
  };

  const size_t MT = (size_t)T_DIM * B_DIM;   // 16384 token rows
  const size_t MS = (size_t)S_DIM * B_DIM;   // 4096 memory rows

  __bf16* w_qkv = (__bf16*)alloc((size_t)3 * D_DIM * D_DIM * 2);
  __bf16* w_out = (__bf16*)alloc((size_t)D_DIM * D_DIM * 2);
  __bf16* w_l1  = (__bf16*)alloc((size_t)DFF_D * D_DIM * 2);
  __bf16* w_l2  = (__bf16*)alloc((size_t)D_DIM * DFF_D * 2);
  float*  qn    = (float*)alloc((size_t)NQ_D * D_DIM * 4);
  float*  x1f   = (float*)alloc(MT * D_DIM * 4);
  __bf16* x1b   = (__bf16*)alloc(MT * D_DIM * 2);
  __bf16* memb  = (__bf16*)alloc(MS * D_DIM * 2);
  __bf16* Qb    = (__bf16*)alloc(MT * D_DIM * 2);
  __bf16* Kb    = (__bf16*)alloc(MS * D_DIM * 2);
  __bf16* Vb    = (__bf16*)alloc(MS * D_DIM * 2);
  __bf16* VT    = (__bf16*)alloc(MS * D_DIM * 2);
  __bf16* Ob    = (__bf16*)alloc(MT * D_DIM * 2);
  float*  scr   = (float*)alloc(MT * D_DIM * 4);
  float*  x2f   = (float*)alloc(MT * D_DIM * 4);
  __bf16* x2b   = (__bf16*)alloc(MT * D_DIM * 2);
  __bf16* hb    = (__bf16*)alloc(MT * DFF_D * 2);

  cvt_bf16_kernel<<<1024, 256, 0, stream>>>(in_w, w_qkv, (long long)3 * D_DIM * D_DIM);
  cvt_bf16_kernel<<<512, 256, 0, stream>>>(out_w, w_out, (long long)D_DIM * D_DIM);
  cvt_bf16_kernel<<<1024, 256, 0, stream>>>(l1w, w_l1, (long long)DFF_D * D_DIM);
  cvt_bf16_kernel<<<1024, 256, 0, stream>>>(l2w, w_l2, (long long)D_DIM * DFF_D);
  cvt_bf16_kernel<<<1024, 256, 0, stream>>>(memory, memb, (long long)(MS * D_DIM));

  prep_q_kernel<<<NQ_D * H_DIM, 64, 0, stream>>>(queries, qn);
  const size_t qa_lds = (size_t)(48 * D_DIM + NQ_D * 48 + 48 + D_DIM + 256) * 4;
  qa_ln1_kernel<<<dim3(NW, B_DIM), 256, qa_lds, stream>>>(tgt, qn, wk, ln1g, ln1b,
                                                          x1f, x1b);

  gemm_bf16_kernel<<<dim3(MT / 128, D_DIM / 64), 256, 0, stream>>>(
      x1b, w_qkv, in_b, nullptr, Qb, (int)MT, D_DIM, D_DIM, 0);
  gemm_bf16_kernel<<<dim3(MS / 128, D_DIM / 64), 256, 0, stream>>>(
      memb, w_qkv + (size_t)D_DIM * D_DIM, in_b + D_DIM, nullptr, Kb,
      (int)MS, D_DIM, D_DIM, 0);
  gemm_bf16_kernel<<<dim3(MS / 128, D_DIM / 64), 256, 0, stream>>>(
      memb, w_qkv + (size_t)2 * D_DIM * D_DIM, in_b + 2 * D_DIM, nullptr, Vb,
      (int)MS, D_DIM, D_DIM, 0);
  vt_kernel<<<(int)((MS * D_DIM + 255) / 256), 256, 0, stream>>>(Vb, VT);

  const size_t at_lds = (size_t)(64 * 516 + 64 * 8) * 4;
  attn_kernel<<<dim3(T_DIM / 64, B_DIM * H_DIM), 256, at_lds, stream>>>(Qb, Kb, VT, Ob);

  gemm_bf16_kernel<<<dim3(MT / 128, D_DIM / 64), 256, 0, stream>>>(
      Ob, w_out, out_b, scr, nullptr, (int)MT, D_DIM, D_DIM, 0);
  add_ln_kernel<<<(int)MT, 256, 0, stream>>>(x1f, scr, ln2g, ln2b, x2f, x2b);

  gemm_bf16_kernel<<<dim3(MT / 128, DFF_D / 64), 256, 0, stream>>>(
      x2b, w_l1, l1b, nullptr, hb, (int)MT, DFF_D, D_DIM, 1);
  gemm_bf16_kernel<<<dim3(MT / 128, D_DIM / 64), 256, 0, stream>>>(
      hb, w_l2, l2b, scr, nullptr, (int)MT, D_DIM, DFF_D, 0);
  add_ln_kernel<<<(int)MT, 256, 0, stream>>>(x2f, scr, ln3g, ln3b, (float*)d_out,
                                             nullptr);
}
